// AGCNetVLE6_41162966565506
// MI455X (gfx1250) — matrix operand order repaired and checked by reading, not compiled, not run on hardware
//
#include <hip/hip_runtime.h>

// AGCNet forward for MI455X (gfx1250, wave32). All heavy linears run through a
// bf16 WMMA GEMM (v_wmma_f32_16x16x32_bf16, f32 accumulate); activations,
// segment-softmax, scatters and GRU gates stay f32. Scratch comes from d_ws
// (~1.2 GB peak, reused across stages). Params are consumed in jax
// tree_leaves order (sorted dict keys; lists in order) after the 25 arrays.

#define HD   128
#define NAF  36
#define NBF  12
#define FDF  219
#define NFN  131072
#define EFN  262144
#define NMN  16384
#define EMN  32768
#define BSZ  2048

#define ACT_NONE  0
#define ACT_RELU  1
#define ACT_LRELU 2

typedef unsigned short u16;
typedef unsigned int   u32;
typedef __bf16 v16bf __attribute__((ext_vector_type(16)));
typedef float  v8f   __attribute__((ext_vector_type(8)));

static inline int cdiv(int a, int b) { return (a + b - 1) / b; }

// ---------------------------------------------------------------- device utils
__device__ __forceinline__ u16 f2bf(float f) {
  u32 u = __float_as_uint(f);
  u32 r = (u + 0x7FFFu + ((u >> 16) & 1u)) >> 16;   // round-to-nearest-even
  return (u16)r;
}
__device__ __forceinline__ u32 enc_f32(float f) {   // order-preserving f32->u32
  u32 u = __float_as_uint(f);
  return (u & 0x80000000u) ? ~u : (u | 0x80000000u);
}
__device__ __forceinline__ float dec_f32(u32 u) {
  u32 v = (u & 0x80000000u) ? (u & 0x7FFFFFFFu) : ~u;
  return __uint_as_float(v);
}

// ---------------------------------------------------------------- fills
__global__ void fill_f32_kernel(float* p, float v, int n) {
  int i = blockIdx.x * 256 + threadIdx.x;
  if (i < n) p[i] = v;
}
__global__ void fill_u32_kernel(u32* p, u32 v, int n) {
  int i = blockIdx.x * 256 + threadIdx.x;
  if (i < n) p[i] = v;
}

// ---------------------------------------------------------------- packing
// dst[m, 0:Kp] = bf16(src[idx?idx[m]:m, 0:K]) zero-padded to Kp
__global__ void pack_rows_kernel(u16* dst, const float* src, const int* idx,
                                 int M, int K, int Kp) {
  int i = blockIdx.x * 256 + threadIdx.x;
  if (i >= M * Kp) return;
  int m = i / Kp, k = i - m * Kp;
  int rs = idx ? idx[m] : m;
  dst[i] = (k < K) ? f2bf(src[(size_t)rs * K + k]) : (u16)0;
}
// dst[m, off:off+128] = bf16(src[m, 0:128]); dst row stride Kp
__global__ void pack_cols_kernel(u16* dst, const float* src, int off, int M, int Kp) {
  int i = blockIdx.x * 256 + threadIdx.x;
  if (i >= M * HD) return;
  int m = i >> 7, k = i & 127;
  dst[(size_t)m * Kp + off + k] = f2bf(src[i]);
}
// dst[e, 0:256] = bf16(concat(A[aIdx?aIdx[e]:e], Bm[bIdx?bIdx[e]:e]))
__global__ void pack2_gather_kernel(u16* dst, const float* A, const int* aIdx,
                                    const float* Bm, const int* bIdx, int E) {
  int i = blockIdx.x * 256 + threadIdx.x;
  if (i >= E * 256) return;
  int e = i >> 8, k = i & 255;
  float v;
  if (k < HD) { int r = aIdx ? aIdx[e] : e; v = A[(size_t)r * HD + k]; }
  else        { int r = bIdx ? bIdx[e] : e; v = Bm[(size_t)r * HD + (k - HD)]; }
  dst[i] = f2bf(v);
}
// weights: dst[k, n] = bf16(w[k, n]) for k<K else 0 ; dst is [Kp, N]
__global__ void pack_w_kernel(u16* dst, const float* w, int K, int Kp, int N) {
  int i = blockIdx.x * 256 + threadIdx.x;
  if (i >= Kp * N) return;
  int k = i / N, n = i - k * N;
  dst[i] = (k < K) ? f2bf(w[(size_t)k * N + n]) : (u16)0;
}
__global__ void pack_pred_scalars_kernel(u16* dst, const float* tb1, const float* tc1,
                                         const float* tb2, const float* tc2,
                                         const float* xx, int B) {
  int b = blockIdx.x * 256 + threadIdx.x;
  if (b >= B) return;
  size_t base = (size_t)b * 288 + 256;
  dst[base + 0] = f2bf(tb1[b]);
  dst[base + 1] = f2bf(tc1[b]);
  dst[base + 2] = f2bf(tb2[b]);
  dst[base + 3] = f2bf(tc2[b]);
  dst[base + 4] = f2bf(xx[b]);
}

// ---------------------------------------------------------------- WMMA GEMM
// C[M,N] = act(A[M,Kp](bf16) * W[Kp,N](bf16) + bias[N]) , f32 out.
// Block: 256 threads = 8 waves; tile BM=128 x BN=128, K-step 32.
// Wave w computes rows [16w,16w+16) x 128 cols = 8 accumulators (16x16 f32).
// LDS holds bf16 data pre-packed as K-pairs (one u32 = {k,k+1}) so each
// fragment element is a single conflict-free ds_load_b32 (strides 17 / 129).
__global__ __launch_bounds__(256) void gemm_bf16_kernel(
    const u16* __restrict__ A, const u16* __restrict__ W,
    const float* __restrict__ bias, float* __restrict__ C,
    int M, int N, int Kp, int act) {
  __shared__ u32 ldsA[128 * 17];   // [row][kpair], pad to 17
  __shared__ u32 ldsB[16 * 129];   // [kpair][col], pad to 129

  const int tid  = threadIdx.x;
  const int lane = tid & 31;
  const int wv   = tid >> 5;
  const int g    = lane >> 4;      // half-wave group
  const int ml   = lane & 15;
  const int rowBase = blockIdx.x * 128;
  const int colBase = blockIdx.y * 128;

  v8f acc[8];
  for (int i = 0; i < 8; ++i)
    for (int j = 0; j < 8; ++j) acc[i][j] = 0.0f;

  for (int k0 = 0; k0 < Kp; k0 += 32) {
    { // A tile: rows rowBase..+127, k k0..k0+31 (memory pairs are natural)
      int r  = tid >> 1;
      int kh = (tid & 1) * 16;
      int grow = rowBase + r;
      uint4 q0 = make_uint4(0, 0, 0, 0), q1 = q0;
      if (grow < M) {
        const uint4* s = (const uint4*)(A + (size_t)grow * Kp + k0 + kh);
        q0 = s[0]; q1 = s[1];
      }
      u32* d = &ldsA[r * 17 + (kh >> 1)];
      d[0] = q0.x; d[1] = q0.y; d[2] = q0.z; d[3] = q0.w;
      d[4] = q1.x; d[5] = q1.y; d[6] = q1.z; d[7] = q1.w;
    }
    { // B tile: interleave rows 2kp / 2kp+1 into {even,odd} u32 pairs
      int kp = tid >> 4;
      int cg = tid & 15;
      const u16* se = W + (size_t)(k0 + 2 * kp) * N + colBase + cg * 8;
      uint4 qe = *(const uint4*)se;
      uint4 qo = *(const uint4*)(se + N);
      const u16* e16 = (const u16*)&qe;
      const u16* o16 = (const u16*)&qo;
      u32* d = &ldsB[kp * 129 + cg * 8];
#pragma unroll
      for (int j = 0; j < 8; ++j)
        d[j] = (u32)e16[j] | ((u32)o16[j] << 16);
    }
    __syncthreads();

    union { v16bf v; u32 u[8]; } fa;
    { // A fragment 16x32: lane (g,ml) ; VGPR v holds K pair p (ISA layout)
      const u32* ar = &ldsA[(wv * 16 + ml) * 17];
#pragma unroll
      for (int v = 0; v < 8; ++v) {
        int p = ((v >> 2) << 3) + (g << 2) + (v & 3);
        fa.u[v] = ar[p];
      }
    }
#pragma unroll
    for (int nb = 0; nb < 8; ++nb) {
      union { v16bf v; u32 u[8]; } fb;
#pragma unroll
      for (int v = 0; v < 8; ++v)
        fb.u[v] = ldsB[(((v >> 2) << 3) + (g << 2) + (v & 3)) * 129 + nb * 16 + ml];
      acc[nb] = __builtin_amdgcn_wmma_f32_16x16x32_bf16(
          false, fa.v, false, fb.v, (short)0, acc[nb], false, false);
    }
    __syncthreads();
  }

  // C layout: VGPR r -> M = r + 8*g ; lanes give N = ml
#pragma unroll
  for (int nb = 0; nb < 8; ++nb) {
    int col = colBase + nb * 16 + ml;
    float bv = bias ? bias[col] : 0.0f;
#pragma unroll
    for (int r = 0; r < 8; ++r) {
      int row = rowBase + wv * 16 + r + 8 * g;
      if (row < M) {
        float v = acc[nb][r] + bv;
        if (act == ACT_RELU)       v = v > 0.f ? v : 0.f;
        else if (act == ACT_LRELU) v = v > 0.f ? v : 0.01f * v;
        C[(size_t)row * N + col] = v;
      }
    }
  }
}

// ---------------------------------------------------------------- attention dot
// lg[e] = lrelu( dot(A[aIdx?aIdx[e]:e], w[0:128]) +
//                dot(Bm[bIdx?bIdx[e]:e], w[128:256]) + bias[0] )
__global__ __launch_bounds__(256) void att_dot_kernel(
    float* __restrict__ lg, const float* __restrict__ A, const int* aIdx,
    const float* __restrict__ Bm, const int* bIdx,
    const float* __restrict__ w, const float* __restrict__ bias, int E) {
  int wv = threadIdx.x >> 5, lane = threadIdx.x & 31;
  int e = blockIdx.x * 8 + wv;
  if (e >= E) return;
  int ra = aIdx ? aIdx[e] : e;
  int rb = bIdx ? bIdx[e] : e;
  const float* pa = A + (size_t)ra * HD;
  const float* pb = Bm + (size_t)rb * HD;
  float acc = 0.f;
#pragma unroll
  for (int j = 0; j < 4; ++j) {
    int c = lane * 4 + j;
    acc += pa[c] * w[c] + pb[c] * w[HD + c];
  }
  for (int m = 16; m >= 1; m >>= 1) acc += __shfl_xor(acc, m, 32);
  if (lane == 0) {
    float v = acc + bias[0];
    lg[e] = v > 0.f ? v : 0.01f * v;
  }
}

// row-dot output head: out[b] = act(dot(dr[b], w) + bias[0]) ; mode1 = sigmoid
__global__ __launch_bounds__(256) void rowdot_out_kernel(
    float* out, const float* dr, const float* w, const float* bias,
    int mode, int B) {
  int wv = threadIdx.x >> 5, lane = threadIdx.x & 31;
  int b = blockIdx.x * 8 + wv;
  if (b >= B) return;
  const float* p = dr + (size_t)b * HD;
  float acc = 0.f;
#pragma unroll
  for (int j = 0; j < 4; ++j) { int c = lane * 4 + j; acc += p[c] * w[c]; }
  for (int m = 16; m >= 1; m >>= 1) acc += __shfl_xor(acc, m, 32);
  if (lane == 0) {
    float v = acc + bias[0];
    if (mode == 1) v = 1.f / (1.f + expf(-v));
    out[b] = v;
  }
}

// ---------------------------------------------------------------- segment ops
__global__ void segmax_kernel(u32* m, const float* lg, const int* seg, int E) {
  int e = blockIdx.x * 256 + threadIdx.x;
  if (e < E) atomicMax(&m[seg[e]], enc_f32(lg[e]));
}
__global__ void segexp_kernel(float* a, float* s, const float* lg,
                              const u32* m, const int* seg, int E) {
  int e = blockIdx.x * 256 + threadIdx.x;
  if (e >= E) return;
  int d = seg[e];
  float ev = expf(lg[e] - dec_f32(m[d]));
  a[e] = ev;
  atomicAdd(&s[d], ev);
}
__global__ void segnorm_kernel(float* a, const float* s, const int* seg, int E) {
  int e = blockIdx.x * 256 + threadIdx.x;
  if (e < E) a[e] = a[e] / (s[seg[e]] + 1e-9f);
}
// out[seg[e], :] += (a?a[e]:1) * val[e, :]   (4 cols per thread)
__global__ void scatter_wadd_kernel(float* out, const float* a, const float* val,
                                    const int* seg, int E) {
  int idx = blockIdx.x * 256 + threadIdx.x;
  int e = idx >> 5;
  if (e >= E) return;
  int sub = idx & 31;
  int d = seg[e];
  float av = a ? a[e] : 1.0f;
  const float* v = val + (size_t)e * HD + sub * 4;
  float* o = out + (size_t)d * HD + sub * 4;
#pragma unroll
  for (int j = 0; j < 4; ++j) atomicAdd(&o[j], av * v[j]);
}

// ---------------------------------------------------------------- elementwise
__global__ void elu_kernel(float* x, int n) {
  int i = blockIdx.x * 256 + threadIdx.x;
  if (i < n) { float v = x[i]; x[i] = v > 0.f ? v : (expf(v) - 1.f); }
}
__global__ void gru_combine_kernel(float* h, const float* gi, const float* gh, int n) {
  int i = blockIdx.x * 256 + threadIdx.x;
  if (i >= n * HD) return;
  int row = i >> 7, col = i & 127;
  const float* pi = gi + (size_t)row * 384;
  const float* ph = gh + (size_t)row * 384;
  float r  = 1.f / (1.f + expf(-(pi[col] + ph[col])));
  float z  = 1.f / (1.f + expf(-(pi[col + 128] + ph[col + 128])));
  float nn = tanhf(pi[col + 256] + r * ph[col + 256]);
  float ho = h[i];
  h[i] = (1.f - z) * nn + z * ho;
}
__global__ void add_inplace_kernel(float* d, const float* s, int n) {
  int i = blockIdx.x * 256 + threadIdx.x;
  if (i < n) d[i] += s[i];
}
__global__ void scale_relu_kernel(float* o, const float* s, float sc, int n) {
  int i = blockIdx.x * 256 + threadIdx.x;
  if (i < n) { float v = s[i] * sc; o[i] = v > 0.f ? v : 0.f; }
}
__global__ void bn_relu_kernel(float* d, const float* g, const float* bb, int n) {
  int i = blockIdx.x * 256 + threadIdx.x;
  if (i >= n) return;
  int col = i & 127;
  float v = d[i] * g[col] * rsqrtf(1.0f + 1e-5f) + bb[col];
  d[i] = v > 0.f ? v : 0.f;
}

// ================================================================ host side
namespace {

struct Lin { const float *b, *w; };
struct Gru { const float *bh, *bi, *wh, *wi; };
struct ACG { Lin att, ctx; Gru gru; };
struct Atom { Lin att0, ctx0; Gru gru0; Lin ne; ACG l1; };
struct HeadF { Atom atom; ACG mol[2]; };
struct HeadJ { Atom atom; ACG mol[2]; Lin proj; };
struct Comp { Lin emb_fe, emb_fn, emb_me, emb_mn, frag_attend; HeadF fh[4]; HeadJ jh[4]; };

struct Cursor {
  void* const* din; int i;
  const float* next() { return (const float*)din[i++]; }
};
static Lin takeLin(Cursor& c) { Lin l; l.b = c.next(); l.w = c.next(); return l; }
static Gru takeGru(Cursor& c) { Gru g; g.bh = c.next(); g.bi = c.next(); g.wh = c.next(); g.wi = c.next(); return g; }
static ACG takeACG(Cursor& c) { ACG a; a.att = takeLin(c); a.ctx = takeLin(c); a.gru = takeGru(c); return a; }
static Atom takeAtom(Cursor& c) {
  Atom a;
  a.att0 = takeLin(c); a.ctx0 = takeLin(c); a.gru0 = takeGru(c); a.ne = takeLin(c);
  a.l1 = takeACG(c);
  return a;
}
static HeadF takeHeadF(Cursor& c) { HeadF h; h.atom = takeAtom(c); h.mol[0] = takeACG(c); h.mol[1] = takeACG(c); return h; }
static HeadJ takeHeadJ(Cursor& c) { HeadJ h; h.atom = takeAtom(c); h.mol[0] = takeACG(c); h.mol[1] = takeACG(c); h.proj = takeLin(c); return h; }
static Comp takeComp(Cursor& c) {
  Comp p;
  p.emb_fe = takeLin(c); p.emb_fn = takeLin(c); p.emb_me = takeLin(c); p.emb_mn = takeLin(c);
  p.frag_attend = takeLin(c);
  for (int i = 0; i < 4; ++i) p.fh[i] = takeHeadF(c);
  for (int i = 0; i < 4; ++i) p.jh[i] = takeHeadJ(c);
  return p;
}

struct Bufs {
  u16 *Epak, *Npak, *fragcat, *mncat, *predcat, *wpack;
  float *EbufA, *EbufB, *feEmb, *fnEmb, *NodeH, *ctxN, *gi, *gh;
  float *lg, *a, *s, *hgF, *gm, *mnEmb, *meEmb, *xJ, *hgJ, *outacc, *s1, *s2, *dbuf;
  u32 *m;
  hipStream_t st;
};

static void gemm(Bufs& b, const u16* A, const u16* W, const float* bias,
                 float* C, int M, int N, int Kp, int act) {
  dim3 g(cdiv(M, 128), cdiv(N, 128));
  gemm_bf16_kernel<<<g, 256, 0, b.st>>>(A, W, bias, C, M, N, Kp, act);
}
static void packW(Bufs& b, const float* w, int K, int Kp, int N) {
  pack_w_kernel<<<cdiv(Kp * N, 256), 256, 0, b.st>>>(b.wpack, w, K, Kp, N);
}
static void packRows(Bufs& b, u16* dst, const float* src, const int* idx, int M, int K, int Kp) {
  pack_rows_kernel<<<cdiv(M * Kp, 256), 256, 0, b.st>>>(dst, src, idx, M, K, Kp);
}
static void fillF(Bufs& b, float* p, float v, int n) { fill_f32_kernel<<<cdiv(n, 256), 256, 0, b.st>>>(p, v, n); }
static void fillU(Bufs& b, u32* p, u32 v, int n)     { fill_u32_kernel<<<cdiv(n, 256), 256, 0, b.st>>>(p, v, n); }

static void seg_softmax(Bufs& b, float* a, const float* lg, const int* seg, int E, int nseg) {
  fillU(b, b.m, 0u, nseg);
  fillF(b, b.s, 0.f, nseg);
  segmax_kernel<<<cdiv(E, 256), 256, 0, b.st>>>(b.m, lg, seg, E);
  segexp_kernel<<<cdiv(E, 256), 256, 0, b.st>>>(a, b.s, lg, b.m, seg, E);
  segnorm_kernel<<<cdiv(E, 256), 256, 0, b.st>>>(a, b.s, seg, E);
}

static void gru_step(Bufs& b, float* h, const float* ctx, int n, const Gru& G) {
  packRows(b, b.Npak, ctx, nullptr, n, HD, HD);
  packW(b, G.wi, HD, HD, 3 * HD);
  gemm(b, b.Npak, b.wpack, G.bi, b.gi, n, 3 * HD, HD, ACT_NONE);
  packRows(b, b.Npak, h, nullptr, n, HD, HD);
  packW(b, G.wh, HD, HD, 3 * HD);
  gemm(b, b.Npak, b.wpack, G.bh, b.gh, n, 3 * HD, HD, ACT_NONE);
  gru_combine_kernel<<<cdiv(n * HD, 256), 256, 0, b.st>>>(h, b.gi, b.gh, n);
}

static void ctx_scatter(Bufs& b, const float* a, const float* val, const int* seg,
                        int E, int nseg) {
  fillF(b, b.ctxN, 0.f, nseg * HD);
  scatter_wadd_kernel<<<cdiv(E * 32, 256), 256, 0, b.st>>>(b.ctxN, a, val, seg, E);
  elu_kernel<<<cdiv(nseg * HD, 256), 256, 0, b.st>>>(b.ctxN, nseg * HD);
}

static void atom_fp(Bufs& b, float* node, const float* eEmb, const int* src,
                    const int* dst, int E, int n, const Atom& P) {
  // layer 0: edge-conditioned
  pack2_gather_kernel<<<E, 256, 0, b.st>>>(b.Epak, node, src, eEmb, nullptr, E);
  packW(b, P.ne.w, 2 * HD, 2 * HD, HD);
  gemm(b, b.Epak, b.wpack, P.ne.b, b.EbufA, E, HD, 2 * HD, ACT_LRELU);   // he
  att_dot_kernel<<<cdiv(E, 8), 256, 0, b.st>>>(b.lg, node, dst, b.EbufA, nullptr,
                                               P.att0.w, P.att0.b, E);
  seg_softmax(b, b.a, b.lg, dst, E, n);
  packRows(b, b.Epak, b.EbufA, nullptr, E, HD, HD);
  packW(b, P.ctx0.w, HD, HD, HD);
  gemm(b, b.Epak, b.wpack, P.ctx0.b, b.EbufB, E, HD, HD, ACT_NONE);
  ctx_scatter(b, b.a, b.EbufB, dst, E, n);
  gru_step(b, node, b.ctxN, n, P.gru0);
  // layer 1: node-only
  att_dot_kernel<<<cdiv(E, 8), 256, 0, b.st>>>(b.lg, node, dst, node, src,
                                               P.l1.att.w, P.l1.att.b, E);
  seg_softmax(b, b.a, b.lg, dst, E, n);
  packRows(b, b.Epak, node, src, E, HD, HD);
  packW(b, P.l1.ctx.w, HD, HD, HD);
  gemm(b, b.Epak, b.wpack, P.l1.ctx.b, b.EbufB, E, HD, HD, ACT_NONE);
  ctx_scatter(b, b.a, b.EbufB, dst, E, n);
  gru_step(b, node, b.ctxN, n, P.l1.gru);
}

static void mol_fp(Bufs& b, const float* node, const int* gid, int nI, int nG,
                   const ACG* mol, float* hg) {
  fillF(b, hg, 0.f, nG * HD);
  scatter_wadd_kernel<<<cdiv(nI * 32, 256), 256, 0, b.st>>>(hg, nullptr, node, gid, nI);
  packRows(b, b.Epak, node, nullptr, nI, HD, HD);   // node pack reused both steps
  for (int t = 0; t < 2; ++t) {
    att_dot_kernel<<<cdiv(nI, 8), 256, 0, b.st>>>(b.lg, node, nullptr, hg, gid,
                                                  mol[t].att.w, mol[t].att.b, nI);
    seg_softmax(b, b.a, b.lg, gid, nI, nG);
    packW(b, mol[t].ctx.w, HD, HD, HD);
    gemm(b, b.Epak, b.wpack, mol[t].ctx.b, b.EbufB, nI, HD, HD, ACT_NONE);
    ctx_scatter(b, b.a, b.EbufB, gid, nI, nG);
    gru_step(b, hg, b.ctxN, nG, mol[t].gru);
  }
}

static void run_comp(Bufs& b, const Comp& P,
                     const float* fn, const float* fe, const int* fsrc,
                     const int* fdst, const int* fgid,
                     const float* mn, const float* me, const int* msrc,
                     const int* mdst, const int* mgid, float* s_out) {
  // embeddings
  packRows(b, b.Epak, fn, nullptr, NFN, NAF, 64);
  packW(b, P.emb_fn.w, NAF, 64, HD);
  gemm(b, b.Epak, b.wpack, P.emb_fn.b, b.fnEmb, NFN, HD, 64, ACT_RELU);
  packRows(b, b.Epak, fe, nullptr, EFN, NBF, 32);
  packW(b, P.emb_fe.w, NBF, 32, HD);
  gemm(b, b.Epak, b.wpack, P.emb_fe.b, b.feEmb, EFN, HD, 32, ACT_RELU);
  // fragment heads
  for (int h = 0; h < 4; ++h) {
    hipMemcpyAsync(b.NodeH, b.fnEmb, (size_t)NFN * HD * sizeof(float),
                   hipMemcpyDeviceToDevice, b.st);
    atom_fp(b, b.NodeH, b.feEmb, fsrc, fdst, EFN, NFN, P.fh[h].atom);
    mol_fp(b, b.NodeH, fgid, NFN, NMN, P.fh[h].mol, b.hgF);
    pack_cols_kernel<<<cdiv(NMN * HD, 256), 256, 0, b.st>>>(b.fragcat, b.hgF,
                                                            h * HD, NMN, 4 * HD);
  }
  packW(b, P.frag_attend.w, 4 * HD, 4 * HD, HD);
  gemm(b, b.fragcat, b.wpack, P.frag_attend.b, b.gm, NMN, HD, 4 * HD, ACT_RELU);
  // motif embeddings
  packRows(b, b.Epak, mn, nullptr, NMN, FDF, 224);
  packW(b, P.emb_mn.w, FDF, 224, HD);
  gemm(b, b.Epak, b.wpack, P.emb_mn.b, b.mnEmb, NMN, HD, 224, ACT_RELU);
  packRows(b, b.Epak, me, nullptr, EMN, NBF, 32);
  packW(b, P.emb_me.w, NBF, 32, HD);
  gemm(b, b.Epak, b.wpack, P.emb_me.b, b.meEmb, EMN, HD, 32, ACT_RELU);
  // mn = concat(gm, mnEmb)
  pack_cols_kernel<<<cdiv(NMN * HD, 256), 256, 0, b.st>>>(b.mncat, b.gm, 0, NMN, 2 * HD);
  pack_cols_kernel<<<cdiv(NMN * HD, 256), 256, 0, b.st>>>(b.mncat, b.mnEmb, HD, NMN, 2 * HD);
  // junction heads
  fillF(b, b.outacc, 0.f, BSZ * HD);
  for (int h = 0; h < 4; ++h) {
    packW(b, P.jh[h].proj.w, 2 * HD, 2 * HD, HD);
    gemm(b, b.mncat, b.wpack, P.jh[h].proj.b, b.xJ, NMN, HD, 2 * HD, ACT_NONE);
    atom_fp(b, b.xJ, b.meEmb, msrc, mdst, EMN, NMN, P.jh[h].atom);
    mol_fp(b, b.xJ, mgid, NMN, BSZ, P.jh[h].mol, b.hgJ);
    add_inplace_kernel<<<cdiv(BSZ * HD, 256), 256, 0, b.st>>>(b.outacc, b.hgJ, BSZ * HD);
  }
  scale_relu_kernel<<<cdiv(BSZ * HD, 256), 256, 0, b.st>>>(s_out, b.outacc, 0.25f, BSZ * HD);
}

} // namespace

extern "C" void kernel_launch(void* const* d_in, const int* in_sizes, int n_in,
                              void* d_out, int out_size, void* d_ws, size_t ws_size,
                              hipStream_t stream) {
  (void)in_sizes; (void)n_in; (void)out_size; (void)ws_size;

  // ---------------- array inputs (setup_inputs insertion order) ----------------
  const float* fn1   = (const float*)d_in[0];
  const float* fe1   = (const float*)d_in[1];
  const int*   fsrc1 = (const int*)d_in[2];
  const int*   fdst1 = (const int*)d_in[3];
  const int*   fgid1 = (const int*)d_in[4];
  const float* mn1   = (const float*)d_in[5];
  const float* me1   = (const float*)d_in[6];
  const int*   msrc1 = (const int*)d_in[7];
  const int*   mdst1 = (const int*)d_in[8];
  const int*   mgid1 = (const int*)d_in[9];
  const float* fn2   = (const float*)d_in[10];
  const float* fe2   = (const float*)d_in[11];
  const int*   fsrc2 = (const int*)d_in[12];
  const int*   fdst2 = (const int*)d_in[13];
  const int*   fgid2 = (const int*)d_in[14];
  const float* mn2   = (const float*)d_in[15];
  const float* me2   = (const float*)d_in[16];
  const int*   msrc2 = (const int*)d_in[17];
  const int*   mdst2 = (const int*)d_in[18];
  const int*   mgid2 = (const int*)d_in[19];
  const float* tb1   = (const float*)d_in[20];
  const float* tc1   = (const float*)d_in[21];
  const float* tb2   = (const float*)d_in[22];
  const float* tc2   = (const float*)d_in[23];
  const float* xx    = (const float*)d_in[24];

  // ---------------- params: jax tree_leaves order (sorted keys) ---------------
  Cursor c{d_in, 25};
  const float* bn_b = c.next();
  const float* bn_g = c.next();
  Comp c1 = takeComp(c);
  Comp c2 = takeComp(c);
  Lin meanL = takeLin(c);
  Lin predL = takeLin(c);
  Lin varL  = takeLin(c);

  // ---------------- workspace carve-out ---------------------------------------
  size_t off = 0;
  auto take = [&](size_t bytes) -> void* {
    void* r = (char*)d_ws + off;
    off += (bytes + 255) & ~(size_t)255;
    return r;
  };
  Bufs b;
  b.st      = stream;
  b.Epak    = (u16*)take((size_t)EFN * 256 * 2);
  b.EbufA   = (float*)take((size_t)EFN * HD * 4);
  b.EbufB   = (float*)take((size_t)EFN * HD * 4);
  b.feEmb   = (float*)take((size_t)EFN * HD * 4);
  b.fnEmb   = (float*)take((size_t)NFN * HD * 4);
  b.NodeH   = (float*)take((size_t)NFN * HD * 4);
  b.ctxN    = (float*)take((size_t)NFN * HD * 4);
  b.gi      = (float*)take((size_t)NFN * 384 * 4);
  b.gh      = (float*)take((size_t)NFN * 384 * 4);
  b.Npak    = (u16*)take((size_t)NFN * HD * 2);
  b.lg      = (float*)take((size_t)EFN * 4);
  b.a       = (float*)take((size_t)EFN * 4);
  b.m       = (u32*)take((size_t)NFN * 4);
  b.s       = (float*)take((size_t)NFN * 4);
  b.hgF     = (float*)take((size_t)NMN * HD * 4);
  b.fragcat = (u16*)take((size_t)NMN * 512 * 2);
  b.gm      = (float*)take((size_t)NMN * HD * 4);
  b.mnEmb   = (float*)take((size_t)NMN * HD * 4);
  b.meEmb   = (float*)take((size_t)EMN * HD * 4);
  b.mncat   = (u16*)take((size_t)NMN * 256 * 2);
  b.xJ      = (float*)take((size_t)NMN * HD * 4);
  b.hgJ     = (float*)take((size_t)BSZ * HD * 4);
  b.outacc  = (float*)take((size_t)BSZ * HD * 4);
  b.s1      = (float*)take((size_t)BSZ * HD * 4);
  b.s2      = (float*)take((size_t)BSZ * HD * 4);
  b.predcat = (u16*)take((size_t)BSZ * 288 * 2);
  b.dbuf    = (float*)take((size_t)BSZ * HD * 4);
  b.wpack   = (u16*)take((size_t)512 * 384 * 2);

  // ---------------- two component towers --------------------------------------
  run_comp(b, c1, fn1, fe1, fsrc1, fdst1, fgid1, mn1, me1, msrc1, mdst1, mgid1, b.s1);
  run_comp(b, c2, fn2, fe2, fsrc2, fdst2, fgid2, mn2, me2, msrc2, mdst2, mgid2, b.s2);

  // ---------------- prediction head -------------------------------------------
  fillU(b, (u32*)b.predcat, 0u, BSZ * 288 / 2);
  pack_cols_kernel<<<cdiv(BSZ * HD, 256), 256, 0, stream>>>(b.predcat, b.s1, 0, BSZ, 288);
  pack_cols_kernel<<<cdiv(BSZ * HD, 256), 256, 0, stream>>>(b.predcat, b.s2, HD, BSZ, 288);
  pack_pred_scalars_kernel<<<cdiv(BSZ, 256), 256, 0, stream>>>(b.predcat, tb1, tc1,
                                                               tb2, tc2, xx, BSZ);
  packW(b, predL.w, 2 * HD + 5, 288, HD);
  gemm(b, b.predcat, b.wpack, predL.b, b.dbuf, BSZ, HD, 288, ACT_NONE);
  bn_relu_kernel<<<cdiv(BSZ * HD, 256), 256, 0, stream>>>(b.dbuf, bn_g, bn_b, BSZ * HD);

  float* out = (float*)d_out;
  rowdot_out_kernel<<<cdiv(BSZ, 8), 256, 0, stream>>>(out, b.dbuf, meanL.w, meanL.b, 1, BSZ);
  rowdot_out_kernel<<<cdiv(BSZ, 8), 256, 0, stream>>>(out + BSZ, b.dbuf, varL.w, varL.b, 0, BSZ);
}
